// Attention_36369783063344
// MI455X (gfx1250) — matrix operand order repaired and checked by reading, not compiled, not run on hardware
//
#include <hip/hip_runtime.h>
#include <stdint.h>

typedef __attribute__((ext_vector_type(16))) _Float16 v16h;
typedef __attribute__((ext_vector_type(8)))  _Float16 v8h;
typedef __attribute__((ext_vector_type(4)))  _Float16 v4h;
typedef __attribute__((ext_vector_type(8)))  float    v8f;
typedef __attribute__((ext_vector_type(4)))  float    v4f;
typedef __attribute__((ext_vector_type(4)))  uint32_t v4u;
typedef __attribute__((ext_vector_type(8)))  int      v8i;
typedef __attribute__((ext_vector_type(4)))  int      v4i;

#define B_DIM   2
#define H_DIM   16
#define S_LEN   2048
#define D_DIM   64
#define QT      64                 // query rows per workgroup (4 waves x 16)
#define KT      64                 // keys per inner iteration
#define NKB     (S_LEN / KT)
#define LOG2E   1.4426950408889634f
#define SC2     (0.125f * LOG2E)   // (1/sqrt(D)) * log2(e), folded into Q
#define NEGB    (-1.0e9f * LOG2E)  // mask bias in base-2 domain

// f16 workspace layout (bytes): Qh | Kh | Vt   (each B*H*S*D*2 = 8 MB)
#define WS_HALF   ((size_t)B_DIM * H_DIM * S_LEN * D_DIM * 2)

// dynamic-LDS layout (bytes) of the attention kernel (no static __shared__,
// so dynamic LDS starts at offset 0 -> these are also the TDM lds_addr values)
#define SQ_OFF    0u            // 8 KB  [qrow][d]
#define SK_OFF(i) (8192u  + (i)*8192u)   // 2 x 8 KB  [key][d]
#define SVT_OFF(i)(24576u + (i)*8192u)   // 2 x 8 KB  [d][key]
#define SP_OFF    40960u        // 8 KB  per-wave P tiles
#define SBIAS_OFF 49152u        // 8 KB  float bias
#define SMEM_BYTES 57344u

// Build a v16h A/B operand from two contiguous 8-half LDS chunks.
__device__ __forceinline__ v16h ld16(const _Float16* p0, const _Float16* p1) {
    v8h lo = *(const v8h*)p0;
    v8h hi = *(const v8h*)p1;
    return __builtin_shufflevector(lo, hi, 0,1,2,3,4,5,6,7,8,9,10,11,12,13,14,15);
}

#define WMMA_F16(A, Bm, C) \
    __builtin_amdgcn_wmma_f32_16x16x32_f16(false, (A), false, (Bm), (short)0, (C), false, false)

// ---- TDM: async-load a 64x64 f16 tile (row stride `stride0` elements) to LDS ----
__device__ __forceinline__ void tdm_load_tile(uint32_t ldsOff, const _Float16* gptr,
                                              uint32_t stride0) {
    const uint64_t ga = (uint64_t)(uintptr_t)gptr;
    // D# group0: count=1, lds_addr, 57-bit global_addr, type=2
    v4u g0 = { 1u, ldsOff, (uint32_t)ga,
               (uint32_t)((ga >> 32) & 0x01FFFFFFu) | (2u << 30) };
    // D# group1: wg_mask=0, data_size=1(2B); tensor_dim0=64, tensor_dim1=64,
    // tile_dim0=64, tile_dim1=64, tile_dim2=0, stride0, stride1=0
    v8i g1 = { (int)(1u << 16),
               (int)(64u << 16),        // [47:32]=0 | tensor_dim0 lo16
               (int)(64u << 16),        // tensor_dim0 hi16=0 | tensor_dim1 lo16
               (int)(64u << 16),        // tensor_dim1 hi16=0 | tile_dim0=64
               (int)64,                 // tile_dim1=64 | tile_dim2=0
               (int)stride0,            // tensor_dim0_stride lo32
               0, 0 };
    v4i g2 = { 0, 0, 0, 0 };
    v4i g3 = { 0, 0, 0, 0 };
#if defined(__clang_major__) && (__clang_major__ >= 23)
    v8i g4 = { 0, 0, 0, 0, 0, 0, 0, 0 };
    __builtin_amdgcn_tensor_load_to_lds(g0, g1, g2, g3, g4, 0);
#else
    __builtin_amdgcn_tensor_load_to_lds(g0, g1, g2, g3, 0);
#endif
}

// =====================================================================
// Prep: Q*scale -> f16, K -> f16, V -> f16 transposed per head [d][S]
// =====================================================================
__global__ __launch_bounds__(128) void prep_kernel(
        const float* __restrict__ Q, const float* __restrict__ K,
        const float* __restrict__ V, _Float16* __restrict__ Qh,
        _Float16* __restrict__ Kh, _Float16* __restrict__ Vt) {
    __shared__ alignas(16) _Float16 sT[D_DIM * 64];   // transpose staging

    const int tid = threadIdx.x;
    const int j0  = blockIdx.x * 64;      // row (seq) tile
    const int bh  = blockIdx.y;
    const size_t base = (size_t)bh * S_LEN * D_DIM + (size_t)j0 * D_DIM;

    const v4f* q4 = (const v4f*)(Q + base);
    const v4f* k4 = (const v4f*)(K + base);
    const v4f* v4 = (const v4f*)(V + base);
    for (int i = tid; i < 64 * D_DIM / 4; i += 128) {
        v4f qv = q4[i], kv = k4[i], vv = v4[i];
        v4h qh = { (_Float16)(qv[0]*SC2), (_Float16)(qv[1]*SC2),
                   (_Float16)(qv[2]*SC2), (_Float16)(qv[3]*SC2) };
        ((v4h*)(Qh + base))[i] = qh;
        v4h kh = { (_Float16)kv[0], (_Float16)kv[1],
                   (_Float16)kv[2], (_Float16)kv[3] };
        ((v4h*)(Kh + base))[i] = kh;
        const int row  = (i * 4) >> 6;    // seq index in tile
        const int colb = (i * 4) & 63;    // d base
        sT[(colb+0)*64 + row] = (_Float16)vv[0];
        sT[(colb+1)*64 + row] = (_Float16)vv[1];
        sT[(colb+2)*64 + row] = (_Float16)vv[2];
        sT[(colb+3)*64 + row] = (_Float16)vv[3];
    }
    __syncthreads();
    const size_t vtbase = (size_t)bh * D_DIM * S_LEN;
    for (int i = tid; i < 64 * D_DIM / 4; i += 128) {
        const int d  = i >> 4;
        const int kg = (i & 15) * 4;
        *(v4h*)(Vt + vtbase + (size_t)d * S_LEN + j0 + kg) =
            *(const v4h*)(sT + d * 64 + kg);
    }
}

// =====================================================================
// Flash attention: TDM-staged f16 tiles, double-buffered K/V
// =====================================================================
__global__ __launch_bounds__(128) void fattn_kernel(
        const _Float16* __restrict__ Qh, const _Float16* __restrict__ Kh,
        const _Float16* __restrict__ Vt, const uint8_t* __restrict__ M,
        float* __restrict__ O) {
    extern __shared__ char smem[];
    _Float16* sQ    = (_Float16*)(smem + SQ_OFF);
    _Float16* sP    = (_Float16*)(smem + SP_OFF);
    float*    sBias = (float*)(smem + SBIAS_OFF);

    const int tid    = threadIdx.x;
    const int wid    = tid >> 5;
    const int lane   = tid & 31;
    const int halfId = lane >> 4;
    const int lm     = lane & 15;

    const int qtile = blockIdx.x;
    const int bh    = blockIdx.y;
    const int b     = bh / H_DIM;

    const size_t headBase = (size_t)bh * S_LEN * D_DIM;
    const size_t vtHead   = (size_t)bh * D_DIM * S_LEN;
    const uint8_t* mg = M + (size_t)b * S_LEN;

    // ---- TDM: Q tile + first K/V pair; stage mask bias meanwhile ----
    if (wid == 0 && lane == 0) {
        tdm_load_tile(SQ_OFF, Qh + headBase + (size_t)qtile * QT * D_DIM, D_DIM);
        tdm_load_tile(SK_OFF(0),  Kh + headBase, D_DIM);
        tdm_load_tile(SVT_OFF(0), Vt + vtHead,   S_LEN);
    }
    for (int i = tid; i < S_LEN; i += 128)
        sBias[i] = mg[i] ? NEGB : 0.0f;
    if (wid == 0) __builtin_amdgcn_s_wait_tensorcnt(0);
    __syncthreads();

    // ---- running state (C-layout: 8 rows per lane) ----
    float mrun[8], lrun[8];
    v8f   oacc[4];
    #pragma unroll
    for (int t = 0; t < 4; ++t) oacc[t] = (v8f){};
    #pragma unroll
    for (int r = 0; r < 8; ++r) { mrun[r] = -3.0e38f; lrun[r] = 0.0f; }

    const int qrow = wid * 16;

    for (int kb = 0; kb < NKB; ++kb) {
        const int j0  = kb * KT;
        const int cur = kb & 1;

        // issue next pair into the buffer everyone just finished reading,
        // then wait only for the current pair -> DMA overlaps compute
        const bool havePrefetch = (kb + 1) < NKB;
        if (wid == 0 && lane == 0 && havePrefetch) {
            const int nxt = (kb + 1) & 1;
            tdm_load_tile(SK_OFF(nxt),  Kh + headBase + (size_t)(j0 + KT) * D_DIM, D_DIM);
            tdm_load_tile(SVT_OFF(nxt), Vt + vtHead + j0 + KT, S_LEN);
        }
        if (wid == 0) {
            if (havePrefetch) __builtin_amdgcn_s_wait_tensorcnt(2);
            else              __builtin_amdgcn_s_wait_tensorcnt(0);
        }
        __syncthreads();

        const _Float16* sK  = (const _Float16*)(smem + SK_OFF(cur));
        const _Float16* sVt = (const _Float16*)(smem + SVT_OFF(cur));

        // ---- S = Q * K^T : 4 tiles of 16x16 (64 keys), base-2 domain ----
        v8f sc[4];
        #pragma unroll
        for (int t = 0; t < 4; ++t) sc[t] = (v8f){};
        #pragma unroll
        for (int kd = 0; kd < D_DIM; kd += 32) {
            const int m = qrow + lm;
            v16h a = ld16(&sQ[m*D_DIM + kd + halfId*8],
                          &sQ[m*D_DIM + kd + 16 + halfId*8]);
            #pragma unroll
            for (int t = 0; t < 4; ++t) {
                v16h bk = ld16(&sK[(t*16 + lm)*D_DIM + kd + halfId*8],
                               &sK[(t*16 + lm)*D_DIM + kd + 16 + halfId*8]);
                sc[t] = WMMA_F16(a, bk, sc[t]);
            }
        }

        // ---- additive mask bias + online softmax (exp2 domain) ----
        float biasv[4];
        #pragma unroll
        for (int t = 0; t < 4; ++t) biasv[t] = sBias[j0 + t*16 + lm];

        float p[4][8], rmax[8], rsum[8];
        #pragma unroll
        for (int r = 0; r < 8; ++r) {
            #pragma unroll
            for (int t = 0; t < 4; ++t) p[t][r] = sc[t][r] + biasv[t];
            rmax[r] = fmaxf(fmaxf(p[0][r], p[1][r]), fmaxf(p[2][r], p[3][r]));
        }
        #pragma unroll
        for (int xm = 1; xm < 16; xm <<= 1) {
            #pragma unroll
            for (int r = 0; r < 8; ++r)
                rmax[r] = fmaxf(rmax[r], __shfl_xor(rmax[r], xm, 32));
        }
        #pragma unroll
        for (int r = 0; r < 8; ++r) {
            const float mnew  = fmaxf(mrun[r], rmax[r]);
            const float alpha = exp2f(mrun[r] - mnew);
            #pragma unroll
            for (int t = 0; t < 4; ++t) p[t][r] = exp2f(p[t][r] - mnew);
            rsum[r] = (p[0][r] + p[1][r]) + (p[2][r] + p[3][r]);
            mrun[r] = mnew;
            lrun[r] *= alpha;
            oacc[0][r] *= alpha; oacc[1][r] *= alpha;
            oacc[2][r] *= alpha; oacc[3][r] *= alpha;
        }
        #pragma unroll
        for (int xm = 1; xm < 16; xm <<= 1) {
            #pragma unroll
            for (int r = 0; r < 8; ++r)
                rsum[r] += __shfl_xor(rsum[r], xm, 32);
        }
        #pragma unroll
        for (int r = 0; r < 8; ++r) lrun[r] += rsum[r];

        // ---- P: C-layout -> A-layout via per-wave LDS (DS in-order per wave) ----
        _Float16* pw = sP + wid * 16 * KT;
        #pragma unroll
        for (int r = 0; r < 8; ++r) {
            const int row = r + 8 * halfId;
            #pragma unroll
            for (int t = 0; t < 4; ++t)
                pw[row*KT + t*16 + lm] = (_Float16)p[t][r];
        }

        // ---- O += P * V : 4 d-tiles x 2 K-chunks of 32 keys ----
        #pragma unroll
        for (int kc = 0; kc < KT; kc += 32) {
            v16h pa = ld16(&pw[lm*KT + kc + halfId*8],
                           &pw[lm*KT + kc + 16 + halfId*8]);
            #pragma unroll
            for (int t = 0; t < 4; ++t) {
                const int dcol = t * 16 + lm;
                v16h bv = ld16(&sVt[dcol*KT + kc + halfId*8],
                               &sVt[dcol*KT + kc + 16 + halfId*8]);
                oacc[t] = WMMA_F16(pa, bv, oacc[t]);
            }
        }
        __syncthreads();   // everyone done with buffer `cur` before it is re-issued
    }

    // ---- epilogue: O / l ----
    float* og = O + headBase + (size_t)qtile * QT * D_DIM;
    #pragma unroll
    for (int r = 0; r < 8; ++r) {
        const float inv = 1.0f / lrun[r];
        const int row = qrow + r + 8 * halfId;
        #pragma unroll
        for (int t = 0; t < 4; ++t)
            og[row * D_DIM + t*16 + lm] = oacc[t][r] * inv;
    }
}

extern "C" void kernel_launch(void* const* d_in, const int* in_sizes, int n_in,
                              void* d_out, int out_size, void* d_ws, size_t ws_size,
                              hipStream_t stream) {
    (void)in_sizes; (void)n_in; (void)ws_size; (void)out_size;
    const float*   q = (const float*)d_in[0];
    const float*   k = (const float*)d_in[1];
    const float*   v = (const float*)d_in[2];
    const uint8_t* m = (const uint8_t*)d_in[3];   // jnp.bool_ -> 1 byte/elem
    float* out = (float*)d_out;

    char* ws = (char*)d_ws;
    _Float16* Qh = (_Float16*)(ws);
    _Float16* Kh = (_Float16*)(ws + WS_HALF);
    _Float16* Vt = (_Float16*)(ws + 2 * WS_HALF);

    dim3 grid(S_LEN / QT, B_DIM * H_DIM);   // (32, 32)
    prep_kernel<<<grid, 128, 0, stream>>>(q, k, v, Qh, Kh, Vt);
    fattn_kernel<<<grid, 128, SMEM_BYTES, stream>>>(Qh, Kh, Vt, m, out);
}
